// MultiHeadAttention2D_CHA_24111946399778
// MI455X (gfx1250) — matrix operand order repaired and checked by reading, not compiled, not run on hardware
//
#include <hip/hip_runtime.h>
#include <hip/hip_bf16.h>
#include <cstdint>
#include <cstddef>

// ---------------------------------------------------------------------------
// MultiHeadAttention2D channel-attention for MI455X (gfx1250, wave32, WMMA).
//
// Shapes: B=8, C=256, H=W=64, HEADS=8, STRIDE=2 -> L = 32*32 = 1024, BH = 64.
// Pipeline (all scratch in d_ws):
//   Pq  bf16 [bh][c][l]   @ off 0          (33,554,432 B)
//   Pk  bf16 [bh][d][l]   @ off 32 MiB     (33,554,432 B)
//   PvT bf16 [bh][l][d]   @ off 64 MiB     (33,554,432 B)
//   attn bf16 [bh][c][d]  @ off 96 MiB     ( 8,388,608 B)
//   O   f32  [bh][c][l]   @ off 104 MiB    (67,108,864 B)   total ~168 MiB
//
// GEMM block tile: M64 x N256, 8 waves in a 2x4 grid, each wave M32 x N64
// (8 f32 WMMA accumulators = 64 VGPRs) -> 8 v_wmma per K-step per wave.
// ---------------------------------------------------------------------------

typedef __attribute__((ext_vector_type(16))) __bf16          v16bf;
typedef __attribute__((ext_vector_type(8)))  float           v8f;
typedef __attribute__((ext_vector_type(8)))  unsigned short  us8;
typedef __attribute__((ext_vector_type(16))) unsigned short  us16;

#define BATCH 8
#define CCH   256
#define NHEAD 8
#define HH    64
#define WWID  64
#define LSP   1024
#define BH    64

__device__ __forceinline__ unsigned short f2bf(float f) {
  unsigned int u = __builtin_bit_cast(unsigned int, f);
  u += 0x7FFFu + ((u >> 16) & 1u);           // round-to-nearest-even
  return (unsigned short)(u >> 16);
}

// ---------------------------------------------------------------------------
// Kernel 1: grouped 2x2/stride-2 conv projections for q and k.
// Conv output channel j (j = h*256 + c after the reference reshape) reads
// input channel j>>3 and weight w[j]. Flat Pq index == tid by construction.
// ---------------------------------------------------------------------------
__global__ __launch_bounds__(256) void proj_qk_kernel(
    const float* __restrict__ q,  const float* __restrict__ k,
    const float* __restrict__ wq, const float* __restrict__ bq,
    const float* __restrict__ wk, const float* __restrict__ bk,
    unsigned short* __restrict__ Pq, unsigned short* __restrict__ Pk) {
  unsigned int tid = blockIdx.x * blockDim.x + threadIdx.x;   // B*2048*1024
  unsigned int l = tid & 1023u;
  unsigned int j = (tid >> 10) & 2047u;
  unsigned int b = tid >> 21;
  unsigned int y = l >> 5, x = l & 31u;
  unsigned int cin = j >> 3;
  size_t ib = (((size_t)(b * CCH + cin)) * HH + 2u * y) * WWID + 2u * x;

  const float* w4q = wq + (size_t)j * 4;
  float pvq = bq[j] + q[ib] * w4q[0] + q[ib + 1] * w4q[1]
                    + q[ib + WWID] * w4q[2] + q[ib + WWID + 1] * w4q[3];
  Pq[tid] = f2bf(pvq);

  const float* w4k = wk + (size_t)j * 4;
  float pvk = bk[j] + k[ib] * w4k[0] + k[ib + 1] * w4k[1]
                    + k[ib + WWID] * w4k[2] + k[ib + WWID + 1] * w4k[3];
  Pk[tid] = f2bf(pvk);
}

// ---------------------------------------------------------------------------
// Kernel 2: v projection, stored transposed as PvT[bh][l][c] so GEMM2's
// B-matrix fragments (K = d) are per-lane contiguous. Flat PvT index == tid.
// ---------------------------------------------------------------------------
__global__ __launch_bounds__(256) void proj_v_kernel(
    const float* __restrict__ v, const float* __restrict__ wv,
    const float* __restrict__ bv, unsigned short* __restrict__ PvT) {
  unsigned int tid = blockIdx.x * blockDim.x + threadIdx.x;   // B*H*L*C
  unsigned int c = tid & 255u;
  unsigned int l = (tid >> 8) & 1023u;
  unsigned int h = (tid >> 18) & 7u;
  unsigned int b = tid >> 21;
  unsigned int j = h * 256u + c;
  unsigned int cin = j >> 3;
  unsigned int y = l >> 5, x = l & 31u;
  size_t ib = (((size_t)(b * CCH + cin)) * HH + 2u * y) * WWID + 2u * x;
  const float* w4 = wv + (size_t)j * 4;
  float pv = bv[j] + v[ib] * w4[0] + v[ib + 1] * w4[1]
                   + v[ib + WWID] * w4[2] + v[ib + WWID + 1] * w4[3];
  PvT[tid] = f2bf(pv);
}

// ---------------------------------------------------------------------------
// WMMA fragment loaders in contiguous k order, not the ISA table order (16-bit, wave32).
// A 16x32: lane m (<16) holds K = 0..15; lane m+16 holds K = 16..31
// of row m.  B 32x16: lane n (<16) holds K = 0..15 of column n; lane
// n+16 holds K = 16..31 of column n.
// ---------------------------------------------------------------------------
__device__ __forceinline__ v16bf load_a_frag(const unsigned short* rowPtr,
                                             int k0, int lhalf) {
  const unsigned short* p = rowPtr + k0 + lhalf * 16;
  us8 lo = *(const us8*)(p);
  us8 hi = *(const us8*)(p + 8);
  us16 u = __builtin_shufflevector(lo, hi, 0, 1, 2, 3, 4, 5, 6, 7,
                                           8, 9, 10, 11, 12, 13, 14, 15);
  union { us16 u; v16bf v; } cvt; cvt.u = u;
  return cvt.v;
}

__device__ __forceinline__ v16bf load_b_frag(const unsigned short* colPtr,
                                             int k0, int lhalf) {
  union { us16 u; v16bf v; } cvt;
  cvt.u = *(const us16*)(colPtr + k0 + lhalf * 16);
  return cvt.v;
}

#define WMMA_BF16(A, B, Cm) \
  __builtin_amdgcn_wmma_f32_16x16x32_bf16(false, (A), false, (B), (short)0, (Cm), false, false)

// ---------------------------------------------------------------------------
// Kernel 3: S = (Pq @ Pk^T) * L^-0.5 ; softmax over d ; attn bf16 out.
// Grid: (4 M-blocks of 64 rows, 64 bh). Block: 256 threads = 8 waves in a
// 2x4 grid; wave (wrow,wcol) computes rows [r0+wrow*32, +32) x cols
// [wcol*64, +64): 8 WMMA tiles, K-loop of 1024 (256 v_wmma per wave).
// ---------------------------------------------------------------------------
__global__ __launch_bounds__(256) void gemm1_softmax_kernel(
    const unsigned short* __restrict__ Pq, const unsigned short* __restrict__ Pk,
    unsigned short* __restrict__ attn) {
  const int bh   = blockIdx.y;
  const int r0   = blockIdx.x * 64;
  const int wave = threadIdx.x >> 5;
  const int wrow = wave >> 2;          // 0..1
  const int wcol = wave & 3;           // 0..3
  const int lane = threadIdx.x & 31;
  const int lhalf = lane >> 4;
  const int llow  = lane & 15;

  __shared__ float S[64][264];
  __shared__ float rmax_s[64];
  __shared__ float rinv_s[64];

  v8f acc[2][4] = {{{}, {}, {}, {}}, {{}, {}, {}, {}}};

  const unsigned short* aRow[2];
  const unsigned short* bCol[4];
#pragma unroll
  for (int mt = 0; mt < 2; ++mt)
    aRow[mt] = Pq + ((size_t)(bh * 256 + r0 + wrow * 32 + mt * 16 + llow) << 10);
#pragma unroll
  for (int nt = 0; nt < 4; ++nt)
    bCol[nt] = Pk + ((size_t)(bh * 256 + wcol * 64 + nt * 16 + llow) << 10);

#pragma unroll 4
  for (int k0 = 0; k0 < 1024; k0 += 32) {
    v16bf a[2], b[4];
#pragma unroll
    for (int mt = 0; mt < 2; ++mt) a[mt] = load_a_frag(aRow[mt], k0, lhalf);
#pragma unroll
    for (int nt = 0; nt < 4; ++nt) b[nt] = load_b_frag(bCol[nt], k0, lhalf);
#pragma unroll
    for (int mt = 0; mt < 2; ++mt)
#pragma unroll
      for (int nt = 0; nt < 4; ++nt)
        acc[mt][nt] = WMMA_BF16(a[mt], b[nt], acc[mt][nt]);
  }

  // C/D layout: element e -> row = e + 8*lhalf, col = llow.
  const float scale = 0.03125f;  // 1024^-0.5
#pragma unroll
  for (int mt = 0; mt < 2; ++mt)
#pragma unroll
    for (int nt = 0; nt < 4; ++nt) {
      int col = wcol * 64 + nt * 16 + llow;
#pragma unroll
      for (int e = 0; e < 8; ++e) {
        int row = wrow * 32 + mt * 16 + e + lhalf * 8;
        S[row][col] = acc[mt][nt][e] * scale;
      }
    }
  __syncthreads();

  if (threadIdx.x < 64) {
    int r = threadIdx.x;
    float m = -3.402823466e38f;
    for (int c = 0; c < 256; ++c) m = fmaxf(m, S[r][c]);
    float s = 0.f;
    for (int c = 0; c < 256; ++c) s += __expf(S[r][c] - m);
    rmax_s[r] = m;
    rinv_s[r] = 1.f / s;
  }
  __syncthreads();

  unsigned short* aout = attn + ((size_t)bh << 16) + (size_t)r0 * 256 + threadIdx.x;
#pragma unroll 4
  for (int i = 0; i < 64; ++i) {
    float vv = __expf(S[i][threadIdx.x] - rmax_s[i]) * rinv_s[i];
    aout[(size_t)i * 256] = f2bf(vv);
  }
}

// ---------------------------------------------------------------------------
// Kernel 4: O = attn @ Pv.  A = attn[bh][c][d] (K=d=256), B via PvT[bh][l][d].
// Grid: (4 M-blocks of 64, 4 N-blocks of 256, 64 bh). Block 256 = 8 waves in
// 2x4 grid, wave tile M32 x N64, K=256 -> 64 v_wmma per wave.
// ---------------------------------------------------------------------------
__global__ __launch_bounds__(256) void gemm2_kernel(
    const unsigned short* __restrict__ attn, const unsigned short* __restrict__ PvT,
    float* __restrict__ O) {
  const int bh = blockIdx.z;
  const int r0 = blockIdx.x * 64;
  const int l0 = blockIdx.y * 256;
  const int wave = threadIdx.x >> 5;
  const int wrow = wave >> 2;
  const int wcol = wave & 3;
  const int lane = threadIdx.x & 31;
  const int lhalf = lane >> 4;
  const int llow  = lane & 15;

  v8f acc[2][4] = {{{}, {}, {}, {}}, {{}, {}, {}, {}}};

  const unsigned short* aRow[2];
  const unsigned short* bCol[4];
#pragma unroll
  for (int mt = 0; mt < 2; ++mt)
    aRow[mt] = attn + ((size_t)bh << 16)
                    + ((size_t)(r0 + wrow * 32 + mt * 16 + llow) << 8);
#pragma unroll
  for (int nt = 0; nt < 4; ++nt)
    bCol[nt] = PvT + ((size_t)bh << 18)
                   + ((size_t)(l0 + wcol * 64 + nt * 16 + llow) << 8);

#pragma unroll
  for (int k0 = 0; k0 < 256; k0 += 32) {
    v16bf a[2], b[4];
#pragma unroll
    for (int mt = 0; mt < 2; ++mt) a[mt] = load_a_frag(aRow[mt], k0, lhalf);
#pragma unroll
    for (int nt = 0; nt < 4; ++nt) b[nt] = load_b_frag(bCol[nt], k0, lhalf);
#pragma unroll
    for (int mt = 0; mt < 2; ++mt)
#pragma unroll
      for (int nt = 0; nt < 4; ++nt)
        acc[mt][nt] = WMMA_BF16(a[mt], b[nt], acc[mt][nt]);
  }

  float* obase = O + ((size_t)bh << 18);
#pragma unroll
  for (int mt = 0; mt < 2; ++mt)
#pragma unroll
    for (int nt = 0; nt < 4; ++nt) {
      int col = l0 + wcol * 64 + nt * 16 + llow;
#pragma unroll
      for (int e = 0; e < 8; ++e) {
        int row = r0 + wrow * 32 + mt * 16 + e + lhalf * 8;
        obase[(size_t)row * 1024 + col] = acc[mt][nt][e];
      }
    }
}

// ---------------------------------------------------------------------------
// Kernel 5: grouped 2x2/stride-2 conv-transpose (non-overlapping) + residual.
// out[b,o,2y+r,2x+s] = q + gamma*(bo[o] + sum_j wo[o,j,1-r,1-s]*O[b, o*8+j, y,x])
// where O channel m = h*256 + c (h = m>>8, c = m&255).
// ---------------------------------------------------------------------------
__global__ __launch_bounds__(256) void deconv_out_kernel(
    const float* __restrict__ O, const float* __restrict__ qin,
    const float* __restrict__ wo, const float* __restrict__ bo,
    const float* __restrict__ gamma, float* __restrict__ out) {
  unsigned int tid = blockIdx.x * blockDim.x + threadIdx.x;   // B*256*1024
  unsigned int l = tid & 1023u;
  unsigned int o = (tid >> 10) & 255u;
  unsigned int b = tid >> 18;
  unsigned int y = l >> 5, x = l & 31u;

  float g = gamma[0];
  float bias = bo[o];
  float a00 = bias, a01 = bias, a10 = bias, a11 = bias;  // [r][s]

#pragma unroll
  for (int j = 0; j < 8; ++j) {
    unsigned int m = o * 8u + j;
    unsigned int h = m >> 8, c = m & 255u;
    float ov = O[((size_t)(b * 8u + h) << 18) + ((size_t)c << 10) + l];
    const float* w = wo + (size_t)m * 4;
    a00 += ov * w[3];   // (r=0,s=0) -> w[1][1]
    a01 += ov * w[2];   // (r=0,s=1) -> w[1][0]
    a10 += ov * w[1];   // (r=1,s=0) -> w[0][1]
    a11 += ov * w[0];   // (r=1,s=1) -> w[0][0]
  }

  size_t base = (((size_t)(b * CCH + o)) * HH + 2u * y) * WWID + 2u * x;
  out[base]            = qin[base]            + g * a00;
  out[base + 1]        = qin[base + 1]        + g * a01;
  out[base + WWID]     = qin[base + WWID]     + g * a10;
  out[base + WWID + 1] = qin[base + WWID + 1] + g * a11;
}

// ---------------------------------------------------------------------------
extern "C" void kernel_launch(void* const* d_in, const int* in_sizes, int n_in,
                              void* d_out, int out_size, void* d_ws, size_t ws_size,
                              hipStream_t stream) {
  const float* q     = (const float*)d_in[0];
  const float* k     = (const float*)d_in[1];
  const float* v     = (const float*)d_in[2];
  const float* wq    = (const float*)d_in[3];
  const float* bq    = (const float*)d_in[4];
  const float* wk    = (const float*)d_in[5];
  const float* bk    = (const float*)d_in[6];
  const float* wv    = (const float*)d_in[7];
  const float* bv    = (const float*)d_in[8];
  const float* wo    = (const float*)d_in[9];
  const float* bo    = (const float*)d_in[10];
  const float* gamma = (const float*)d_in[11];
  float* out = (float*)d_out;

  char* ws = (char*)d_ws;
  unsigned short* Pq   = (unsigned short*)(ws);
  unsigned short* Pk   = (unsigned short*)(ws + (size_t)33554432);
  unsigned short* PvT  = (unsigned short*)(ws + (size_t)67108864);
  unsigned short* attn = (unsigned short*)(ws + (size_t)100663296);
  float*          Obuf = (float*)        (ws + (size_t)109051904);

  proj_qk_kernel<<<65536, 256, 0, stream>>>(q, k, wq, bq, wk, bk, Pq, Pk);
  proj_v_kernel <<<65536, 256, 0, stream>>>(v, wv, bv, PvT);
  gemm1_softmax_kernel<<<dim3(4, 64), 256, 0, stream>>>(Pq, Pk, attn);
  gemm2_kernel<<<dim3(4, 4, 64), 256, 0, stream>>>(attn, PvT, Obuf);
  deconv_out_kernel<<<8192, 256, 0, stream>>>(Obuf, q, wo, bo, gamma, out);
}
